// MaskedContextSelector_31336081391682
// MI455X (gfx1250) — matrix operand order repaired and checked by reading, not compiled, not run on hardware
//
#include <hip/hip_runtime.h>

// ---------------- problem constants ----------------
#define BB 2
#define TT 2048
#define EE 1024
#define HH 16
#define DH 64
#define NEGV (-1000000000.0f)
#define SCALE 0.125f           // 1/sqrt(64)
#define INV_TAU 0.5f
#define NINF (-3.0e38f)

#define AS1 __attribute__((address_space(1)))
#define AS3 __attribute__((address_space(3)))

typedef __bf16 bf16_t;
typedef bf16_t v16bf __attribute__((ext_vector_type(16)));
typedef bf16_t v8bf  __attribute__((ext_vector_type(8)));
typedef float  v8f   __attribute__((ext_vector_type(8)));
typedef int    v4i   __attribute__((ext_vector_type(4)));

// ---------------- WMMA fragment loaders (CDNA5 wave32 layouts) ----------------
// A-matrix 16x32 bf16: lane m = lane%16, half = lane/16,
//   K = {8h..8h+7} U {16+8h..16+8h+7}  -> two 16B loads
__device__ __forceinline__ v16bf load_a_frag(const bf16_t* __restrict__ base, int stride) {
    const int lane = threadIdx.x & 31;
    const int m = lane & 15, hh = lane >> 4;
    const bf16_t* p = base + m * stride + 8 * hh;
    v8bf lo = *(const v8bf*)(p);
    v8bf hi = *(const v8bf*)(p + 16);
    v16bf r;
#pragma unroll
    for (int i = 0; i < 8; ++i) { r[i] = lo[i]; r[8 + i] = hi[i]; }
    return r;
}

// B-matrix 32x16 bf16 fed as B^T rows: lane n = lane%16, half = lane/16,
//   K = {8h..8h+7} U {16+8h..16+8h+7}  -> two 16B loads
__device__ __forceinline__ v16bf load_b_frag(const bf16_t* __restrict__ base, int stride) {
    const int lane = threadIdx.x & 31;
    const int n = lane & 15, hh = lane >> 4;
    const bf16_t* p = base + n * stride + 8 * hh;
    v8bf lo = *(const v8bf*)(p);
    v8bf hi = *(const v8bf*)(p + 16);
    v16bf r;
#pragma unroll
    for (int i = 0; i < 8; ++i) { r[i] = lo[i]; r[8 + i] = hi[i]; }
    return r;
}

__device__ __forceinline__ v8f wmma_bf16(v16bf a, v16bf b, v8f c) {
    return __builtin_amdgcn_wmma_f32_16x16x32_bf16(false, a, false, b, (short)0, c, false, false);
}

// ---------------- CDNA5 async memory->LDS staging (ASYNCcnt path) -----------------
template <int OFF>
__device__ __forceinline__ void async_cp16(const bf16_t* g, bf16_t* l) {
    __builtin_amdgcn_global_load_async_to_lds_b128((AS1 v4i*)g, (AS3 v4i*)l, OFF, 0);
}

// Stage one 128x32 bf16 tile pair (A rows + B^T rows) into LDS (stride 40 elems).
// 256 threads: each moves 2x16B of A and 2x16B of B via async-to-LDS.
__device__ __forceinline__ void stage_tiles_async(const bf16_t* __restrict__ gA, size_t sA,
                                                  const bf16_t* __restrict__ gB, size_t sB,
                                                  bf16_t* lA, bf16_t* lB, int tid) {
    const int row = tid >> 1;              // 0..127
    const int segb = (tid & 1) * 16;       // element offset 0 or 16
    const bf16_t* ga = gA + (size_t)row * sA + segb;
    const bf16_t* gb = gB + (size_t)row * sB + segb;
    bf16_t* la = lA + row * 40 + segb;
    bf16_t* lb = lB + row * 40 + segb;
    async_cp16<0>(ga, la);
    async_cp16<16>(ga, la);                // +8 elements via immediate offset
    async_cp16<0>(gb, lb);
    async_cp16<16>(gb, lb);
}

// ---------------- kernel 1a: pack hidden_states -> bf16 (row major + transposed) ----
__global__ __launch_bounds__(256) void pack_hs_kernel(const float* __restrict__ hs,
                                                      bf16_t* __restrict__ hsb,
                                                      bf16_t* __restrict__ hsT) {
    int idx = blockIdx.x * 256 + threadIdx.x;          // over BB*TT*EE
    int m = idx >> 10;                                  // b*TT + t
    int e = idx & 1023;
    int b = m >> 11;
    int t = m & 2047;
    float v = hs[idx];
    bf16_t bv = (bf16_t)v;
    hsb[idx] = bv;
    hsT[((size_t)(b * EE + e) << 11) + t] = bv;        // [b][e][t], stride TT
}

// ---------------- kernel 1b: pack first 2E rows of in_proj_weight -> bf16 ----------
__global__ __launch_bounds__(256) void pack_w_kernel(const float* __restrict__ W,
                                                     bf16_t* __restrict__ wqk) {
    int idx = blockIdx.x * 256 + threadIdx.x;          // over 2*EE*EE
    wqk[idx] = (bf16_t)W[idx];
}

// ---------------- kernel 2: QK projection GEMM (M=4096, N=2048, K=1024) -----------
// qkb[m][o] = sum_e hsb[m][e] * wqk[o][e] + bias[o]   (o<E -> Q, o>=E -> K)
__global__ __launch_bounds__(256) void qk_gemm_kernel(const bf16_t* __restrict__ hsb,
                                                      const bf16_t* __restrict__ wqk,
                                                      const float* __restrict__ bias,
                                                      bf16_t* __restrict__ qkb) {
    __shared__ alignas(16) bf16_t As[2][128 * 40];
    __shared__ alignas(16) bf16_t Bs[2][128 * 40];
    const int tid = threadIdx.x;
    const int m0 = blockIdx.y * 128, n0 = blockIdx.x * 128;
    const int w = tid >> 5, lane = tid & 31;
    const int wm = (w >> 2) * 64, wn = (w & 3) * 32;

    const bf16_t* gA = hsb + (size_t)m0 * EE;
    const bf16_t* gB = wqk + (size_t)n0 * EE;

    v8f c[4][2];
#pragma unroll
    for (int mi = 0; mi < 4; ++mi)
#pragma unroll
        for (int ni = 0; ni < 2; ++ni)
#pragma unroll
            for (int r = 0; r < 8; ++r) c[mi][ni][r] = 0.0f;

    const int nk = EE / 32;
    stage_tiles_async(gA, EE, gB, EE, As[0], Bs[0], tid);
    for (int k = 0; k < nk; ++k) {
        const int p = k & 1;
        __builtin_amdgcn_s_wait_asynccnt(0);
        __syncthreads();
        if (k + 1 < nk)
            stage_tiles_async(gA + (k + 1) * 32, EE, gB + (k + 1) * 32, EE,
                              As[1 - p], Bs[1 - p], tid);
        v16bf a[4], bq[2];
#pragma unroll
        for (int mi = 0; mi < 4; ++mi) a[mi] = load_a_frag(&As[p][(wm + mi * 16) * 40], 40);
#pragma unroll
        for (int ni = 0; ni < 2; ++ni) bq[ni] = load_b_frag(&Bs[p][(wn + ni * 16) * 40], 40);
#pragma unroll
        for (int mi = 0; mi < 4; ++mi)
#pragma unroll
            for (int ni = 0; ni < 2; ++ni) c[mi][ni] = wmma_bf16(a[mi], bq[ni], c[mi][ni]);
    }

    const int nl = lane & 15, hh = lane >> 4;
#pragma unroll
    for (int ni = 0; ni < 2; ++ni) {
        int n = n0 + wn + ni * 16 + nl;
        float bv = bias[n];
#pragma unroll
        for (int mi = 0; mi < 4; ++mi)
#pragma unroll
            for (int r = 0; r < 8; ++r) {
                int m = m0 + wm + mi * 16 + r + 8 * hh;
                qkb[(size_t)m * (2 * EE) + n] = (bf16_t)(c[mi][ni][r] + bv);
            }
    }
}

// ---------------- kernel 3a: per-head flash pass 1 -> (rowmax, rowsum) ------------
__global__ __launch_bounds__(128) void stats_kernel(const bf16_t* __restrict__ qkb,
                                                    float2* __restrict__ stats) {
    const int tid = threadIdx.x;
    const int bh = blockIdx.x;               // b*16 + h
    const int b = bh >> 4, h = bh & 15;
    const int w = tid >> 5, lane = tid & 31;
    const int i0 = blockIdx.y * 64 + w * 16;
    const int nl = lane & 15, hh = lane >> 4;

    const bf16_t* Qbase = qkb + (size_t)(b * TT + i0) * (2 * EE) + h * DH;
    v16bf a0 = load_a_frag(Qbase, 2 * EE);
    v16bf a1 = load_a_frag(Qbase + 32, 2 * EE);

    float mrun[8], srun[8];
#pragma unroll
    for (int r = 0; r < 8; ++r) { mrun[r] = NINF; srun[r] = 0.0f; }

    for (int j0 = 0; j0 <= i0; j0 += 16) {
        const bf16_t* Kbase = qkb + (size_t)(b * TT + j0) * (2 * EE) + EE + h * DH;
        v16bf b0 = load_b_frag(Kbase, 2 * EE);
        v16bf b1 = load_b_frag(Kbase + 32, 2 * EE);
        v8f c;
#pragma unroll
        for (int r = 0; r < 8; ++r) c[r] = 0.0f;
        c = wmma_bf16(a0, b0, c);
        c = wmma_bf16(a1, b1, c);
#pragma unroll
        for (int r = 0; r < 8; ++r) {
            int i = i0 + r + 8 * hh;
            int j = j0 + nl;
            float s = (j < i) ? c[r] * SCALE : NEGV;
            float tm = s;
#pragma unroll
            for (int d = 1; d < 16; d <<= 1) tm = fmaxf(tm, __shfl_xor(tm, d));
            float nm = fmaxf(mrun[r], tm);
            float p = __expf(s - nm);
#pragma unroll
            for (int d = 1; d < 16; d <<= 1) p += __shfl_xor(p, d);
            srun[r] = srun[r] * __expf(mrun[r] - nm) + p;
            mrun[r] = nm;
        }
    }
    if (nl == 0) {
#pragma unroll
        for (int r = 0; r < 8; ++r) {
            int i = i0 + r + 8 * hh;
            stats[(size_t)(b * HH + h) * TT + i] = make_float2(mrun[r], srun[r]);
        }
    }
}

// ---------------- kernel 3b: recompute scores, head-averaged probs -> avg ---------
__global__ __launch_bounds__(256) void avg_kernel(const bf16_t* __restrict__ qkb,
                                                  const float2* __restrict__ stats,
                                                  float* __restrict__ avg) {
    const int tid = threadIdx.x;
    const int b = blockIdx.z;
    const int i0 = blockIdx.y * 16;
    const int j0 = blockIdx.x * 128 + (tid >> 5) * 16;
    if (j0 > i0) return;                                 // entire 16x16 tile masked
    const int lane = tid & 31;
    const int nl = lane & 15, hh = lane >> 4;

    const bf16_t* Qbase = qkb + (size_t)(b * TT + i0) * (2 * EE);
    const bf16_t* Kbase = qkb + (size_t)(b * TT + j0) * (2 * EE) + EE;

    float acc[8];
#pragma unroll
    for (int r = 0; r < 8; ++r) acc[r] = 0.0f;

    for (int h = 0; h < HH; ++h) {
        v16bf a0 = load_a_frag(Qbase + h * DH, 2 * EE);
        v16bf a1 = load_a_frag(Qbase + h * DH + 32, 2 * EE);
        v16bf b0 = load_b_frag(Kbase + h * DH, 2 * EE);
        v16bf b1 = load_b_frag(Kbase + h * DH + 32, 2 * EE);
        v8f c;
#pragma unroll
        for (int r = 0; r < 8; ++r) c[r] = 0.0f;
        c = wmma_bf16(a0, b0, c);
        c = wmma_bf16(a1, b1, c);
#pragma unroll
        for (int r = 0; r < 8; ++r) {
            int i = i0 + r + 8 * hh;
            int j = j0 + nl;
            if (j < i) {
                float2 st = stats[(size_t)(b * HH + h) * TT + i];
                acc[r] += __expf(c[r] * SCALE - st.x) / st.y;
            }
        }
    }
#pragma unroll
    for (int r = 0; r < 8; ++r) {
        int i = i0 + r + 8 * hh;
        int j = j0 + nl;
        if (j < i) avg[((size_t)(b * TT + i) << 11) + j] = acc[r] * (1.0f / HH);
    }
}

// ---------------- kernel 4: gumbel softmax over each row -> dw (bf16) -------------
__device__ __forceinline__ float gumbel_of(unsigned long long idx) {
    unsigned long long z = idx * 0x9E3779B97F4A7C15ull + 42ull;
    z ^= z >> 30; z *= 0xBF58476D1CE4E5B9ull;
    z ^= z >> 27; z *= 0x94D049BB133111EBull;
    z ^= z >> 31;
    float u = ((float)(z >> 40) + 0.5f) * (1.0f / 16777216.0f);
    u = fminf(fmaxf(u, 1e-6f), 1.0f - 1e-6f);
    return -__logf(-__logf(u));
}

__global__ __launch_bounds__(256) void gumbel_kernel(const float* __restrict__ avg,
                                                     bf16_t* __restrict__ dwb) {
    __shared__ float part[8];
    __shared__ float bcast;
    const int tid = threadIdx.x;
    const int i = blockIdx.x;
    const int b = blockIdx.y;
    const size_t rowbase = ((size_t)(b * TT + i)) << 11;

    float z[8];
#pragma unroll
    for (int k = 0; k < 8; ++k) {
        int j = tid + k * 256;
        float logit = NEGV;
        if (j < i) {
            float a = avg[rowbase + j];
            logit = fminf(fmaxf(a, -40.0f), 40.0f);
        }
        z[k] = (logit + gumbel_of(rowbase + j)) * INV_TAU;
    }
    // block max
    float lm = z[0];
#pragma unroll
    for (int k = 1; k < 8; ++k) lm = fmaxf(lm, z[k]);
#pragma unroll
    for (int d = 1; d < 32; d <<= 1) lm = fmaxf(lm, __shfl_xor(lm, d));
    if ((tid & 31) == 0) part[tid >> 5] = lm;
    __syncthreads();
    if (tid == 0) {
        float v = part[0];
#pragma unroll
        for (int q = 1; q < 8; ++q) v = fmaxf(v, part[q]);
        bcast = v;
    }
    __syncthreads();
    const float mx = bcast;
    float ls = 0.0f;
#pragma unroll
    for (int k = 0; k < 8; ++k) ls += __expf(z[k] - mx);
#pragma unroll
    for (int d = 1; d < 32; d <<= 1) ls += __shfl_xor(ls, d);
    if ((tid & 31) == 0) part[tid >> 5] = ls;
    __syncthreads();
    if (tid == 0) {
        float v = 0.0f;
#pragma unroll
        for (int q = 0; q < 8; ++q) v += part[q];
        bcast = 1.0f / v;
    }
    __syncthreads();
    const float inv = bcast;
#pragma unroll
    for (int k = 0; k < 8; ++k) {
        int j = tid + k * 256;
        dwb[rowbase + j] = (bf16_t)(__expf(z[k] - mx) * inv);
    }
}

// ---------------- kernel 5: out = dw @ hidden_states  (per b: 2048x1024x2048) -----
__global__ __launch_bounds__(256) void out_gemm_kernel(const bf16_t* __restrict__ dwb,
                                                       const bf16_t* __restrict__ hsT,
                                                       float* __restrict__ out) {
    __shared__ alignas(16) bf16_t As[2][128 * 40];
    __shared__ alignas(16) bf16_t Bs[2][128 * 40];
    const int tid = threadIdx.x;
    const int b = blockIdx.z;
    const int m0 = blockIdx.y * 128, n0 = blockIdx.x * 128;
    const int w = tid >> 5, lane = tid & 31;
    const int wm = (w >> 2) * 64, wn = (w & 3) * 32;

    const bf16_t* gA = dwb + (size_t)b * TT * TT + (size_t)m0 * TT;   // [t][j], stride TT
    const bf16_t* gB = hsT + (size_t)b * EE * TT + (size_t)n0 * TT;   // [e][t], stride TT

    v8f c[4][2];
#pragma unroll
    for (int mi = 0; mi < 4; ++mi)
#pragma unroll
        for (int ni = 0; ni < 2; ++ni)
#pragma unroll
            for (int r = 0; r < 8; ++r) c[mi][ni][r] = 0.0f;

    const int kmax = (m0 + 128 < TT) ? (m0 + 128) : TT;   // dw[t][j]==0 for j>=t
    const int nk = kmax / 32;
    stage_tiles_async(gA, TT, gB, TT, As[0], Bs[0], tid);
    for (int k = 0; k < nk; ++k) {
        const int p = k & 1;
        __builtin_amdgcn_s_wait_asynccnt(0);
        __syncthreads();
        if (k + 1 < nk)
            stage_tiles_async(gA + (k + 1) * 32, TT, gB + (k + 1) * 32, TT,
                              As[1 - p], Bs[1 - p], tid);
        v16bf a[4], bq[2];
#pragma unroll
        for (int mi = 0; mi < 4; ++mi) a[mi] = load_a_frag(&As[p][(wm + mi * 16) * 40], 40);
#pragma unroll
        for (int ni = 0; ni < 2; ++ni) bq[ni] = load_b_frag(&Bs[p][(wn + ni * 16) * 40], 40);
#pragma unroll
        for (int mi = 0; mi < 4; ++mi)
#pragma unroll
            for (int ni = 0; ni < 2; ++ni) c[mi][ni] = wmma_bf16(a[mi], bq[ni], c[mi][ni]);
    }

    const int nl = lane & 15, hh = lane >> 4;
#pragma unroll
    for (int ni = 0; ni < 2; ++ni) {
        int n = n0 + wn + ni * 16 + nl;
#pragma unroll
        for (int mi = 0; mi < 4; ++mi)
#pragma unroll
            for (int r = 0; r < 8; ++r) {
                int t = m0 + wm + mi * 16 + r + 8 * hh;
                float v = (t == 0) ? 0.0f : c[mi][ni][r];
                out[((size_t)(b * TT + t) << 10) + n] = v;
            }
    }
}

// ---------------- host side ----------------
extern "C" void kernel_launch(void* const* d_in, const int* in_sizes, int n_in,
                              void* d_out, int out_size, void* d_ws, size_t ws_size,
                              hipStream_t stream) {
    (void)in_sizes; (void)n_in; (void)out_size; (void)ws_size;
    const float* hs   = (const float*)d_in[0];   // (2,2048,1024)
    const float* W    = (const float*)d_in[1];   // (3072,1024)
    const float* bias = (const float*)d_in[2];   // (3072,)
    float* out = (float*)d_out;

    char* ws = (char*)d_ws;
    size_t off = 0;
    bf16_t* hsb  = (bf16_t*)(ws + off); off += (size_t)BB * TT * EE * 2;        // 8 MB
    bf16_t* hsT  = (bf16_t*)(ws + off); off += (size_t)BB * EE * TT * 2;        // 8 MB
    bf16_t* wqk  = (bf16_t*)(ws + off); off += (size_t)2 * EE * EE * 2;         // 4 MB
    bf16_t* qkb  = (bf16_t*)(ws + off); off += (size_t)BB * TT * 2 * EE * 2;    // 16 MB
    float2* stat = (float2*)(ws + off); off += (size_t)BB * HH * TT * 8;        // 0.5 MB
    float*  avg  = (float*)(ws + off);  off += (size_t)BB * TT * TT * 4;        // 32 MB
    bf16_t* dwb  = (bf16_t*)(ws + off); off += (size_t)BB * TT * TT * 2;        // 16 MB

    pack_hs_kernel<<<(BB * TT * EE) / 256, 256, 0, stream>>>(hs, hsb, hsT);
    pack_w_kernel<<<(2 * EE * EE) / 256, 256, 0, stream>>>(W, wqk);

    qk_gemm_kernel<<<dim3((2 * EE) / 128, (BB * TT) / 128), 256, 0, stream>>>(hsb, wqk, bias, qkb);

    stats_kernel<<<dim3(BB * HH, TT / 64), 128, 0, stream>>>(qkb, stat);

    avg_kernel<<<dim3(TT / 128, TT / 16, BB), 256, 0, stream>>>(qkb, stat, avg);

    gumbel_kernel<<<dim3(TT, BB), 256, 0, stream>>>(avg, dwb);

    out_gemm_kernel<<<dim3(EE / 128, TT / 128, BB), 256, 0, stream>>>(dwb, hsT, out);
}
